// MaxMatchesMatcher_12833362280940
// MI455X (gfx1250) — compile-verified, matrix operand order checked by reading, not run on hardware
//
#include <hip/hip_runtime.h>
#include <hip/hip_bf16.h>
#include <stdint.h>

#define D_DIM 256
#define INV_TEMP 20.0f
#define THRESHOLD 0.01f

typedef __attribute__((ext_vector_type(16))) __bf16 v16bf;
typedef __attribute__((ext_vector_type(8)))  float  v8f;

struct U32x8 { uint4 lo, hi; };

static __device__ __forceinline__ unsigned short f32_to_bf16(float f) {
  unsigned int u = __float_as_uint(f);
  u += 0x7FFFu + ((u >> 16) & 1u);   // round-to-nearest-even
  return (unsigned short)(u >> 16);
}

// One shared P formula so equality comparisons are bitwise self-consistent
// across the pmax / count / writeout kernels (mirrors softmax_row * softmax_col).
static __device__ __forceinline__ float dualP(float s, float rm, float rs, float cm, float cs) {
  float ea = __expf(s - rm);
  float eb = __expf(s - cm);
  return (ea / rs) * (eb / cs);
}

// ---------------- normalize rows (fp32 -> unit-norm bf16) ----------------
__global__ void k_normalize(const float* __restrict__ src, unsigned short* __restrict__ dst) {
  __shared__ float red[256];
  int row = blockIdx.x;
  int t = threadIdx.x;
  float x = src[(size_t)row * D_DIM + t];
  red[t] = x * x;
  __syncthreads();
  for (int s = 128; s > 0; s >>= 1) { if (t < s) red[t] += red[t + s]; __syncthreads(); }
  float inv = rsqrtf(red[0]);
  dst[(size_t)row * D_DIM + t] = f32_to_bf16(x * inv);
}

// ---------------- steering: dst[row,e] = sum_d src[row,d] * S[d,e] (fp32) ----------------
__global__ void k_steer(const float* __restrict__ src, const float* __restrict__ S,
                        float* __restrict__ dst) {
  __shared__ float rowv[D_DIM];
  int row = blockIdx.x;
  int t = threadIdx.x;
  rowv[t] = src[(size_t)row * D_DIM + t];
  __syncthreads();
  float acc = 0.f;
#pragma unroll 8
  for (int d = 0; d < D_DIM; ++d)
    acc = fmaf(rowv[d], S[(size_t)d * D_DIM + t], acc);
  dst[(size_t)row * D_DIM + t] = acc;
}

// ---------------- WMMA similarity GEMM: sim = 20 * Anorm @ Bnorm^T ----------------
// 16-bit A-matrix 16x32 fragment striping (ISA 7.12.2): lane L holds row L%16,
// K = (L/16)*8 + {0..7} in VGPRs 0..3 and K = (L/16)*8 + 16 + {0..7} in VGPRs 4..7.
static __device__ __forceinline__ v16bf load_frag16x32(const unsigned short* __restrict__ base,
                                                       int row0, int kk, int lr, int lh) {
  const unsigned short* p = base + (size_t)(row0 + lr) * D_DIM + kk + lh * 8;
  U32x8 t;
  t.lo = *(const uint4*)(p);        // 8 bf16 at K base
  t.hi = *(const uint4*)(p + 16);   // 8 bf16 at K base + 16
  return __builtin_bit_cast(v16bf, t);
}

static __device__ __forceinline__ void sim_gemm_body(const unsigned short* __restrict__ An,
                                                     const unsigned short* __restrict__ Bn,
                                                     float* __restrict__ sim, int N, int M) {
  int b = blockIdx.z;
  const unsigned short* A  = An + (size_t)b * N * D_DIM;
  const unsigned short* Bv = Bn + (size_t)b * M * D_DIM;
  float* S = sim + (size_t)b * N * M;

  int t = threadIdx.x;
  int w = t >> 5, lane = t & 31;
  int lr = lane & 15, lh = lane >> 4;
  int rw = w >> 1, cw = w & 1;                 // 4x2 wave grid over the 128x128 tile
  int i0 = blockIdx.y * 128 + rw * 32;         // this wave: 32 rows x 64 cols
  int j0 = blockIdx.x * 128 + cw * 64;

  v8f acc[2][4];
  v8f z = {0.f, 0.f, 0.f, 0.f, 0.f, 0.f, 0.f, 0.f};
#pragma unroll
  for (int mi = 0; mi < 2; ++mi)
#pragma unroll
    for (int nj = 0; nj < 4; ++nj) acc[mi][nj] = z;

#pragma unroll
  for (int kk = 0; kk < D_DIM; kk += 32) {
    v16bf af[2], bf[4];
#pragma unroll
    for (int mi = 0; mi < 2; ++mi) af[mi] = load_frag16x32(A, i0 + mi * 16, kk, lr, lh);
#pragma unroll
    for (int nj = 0; nj < 4; ++nj) bf[nj] = load_frag16x32(Bv, j0 + nj * 16, kk, lr, lh);
#pragma unroll
    for (int mi = 0; mi < 2; ++mi)
#pragma unroll
      for (int nj = 0; nj < 4; ++nj)
        acc[mi][nj] = __builtin_amdgcn_wmma_f32_16x16x32_bf16(
            false, af[mi], false, bf[nj], (short)0, acc[mi][nj], false, false);
  }

  // C layout: VGPR r, lanes 0-15 -> M=r, lanes 16-31 -> M=r+8; N = lane%16
#pragma unroll
  for (int mi = 0; mi < 2; ++mi)
#pragma unroll
    for (int nj = 0; nj < 4; ++nj)
#pragma unroll
      for (int r = 0; r < 8; ++r) {
        int gi = i0 + mi * 16 + r + lh * 8;
        int gj = j0 + nj * 16 + lr;
        S[(size_t)gi * M + gj] = acc[mi][nj][r] * INV_TEMP;
      }
}

__global__ void k_simgemm(const unsigned short* __restrict__ An,
                          const unsigned short* __restrict__ Bn,
                          float* __restrict__ sim, int N, int M) {
  sim_gemm_body(An, Bn, sim, N, M);
}

__global__ void k_simgemm_sel(const unsigned short* __restrict__ AnAll, size_t slice,
                              const int* __restrict__ bestRot,
                              const unsigned short* __restrict__ Bn,
                              float* __restrict__ sim, int N, int M) {
  sim_gemm_body(AnAll + (size_t)bestRot[0] * slice, Bn, sim, N, M);
}

// ---------------- row softmax stats ----------------
__global__ void k_rowstats(const float* __restrict__ sim, float* __restrict__ rmax,
                           float* __restrict__ rsum, int N, int M) {
  __shared__ float red[256];
  int blk = blockIdx.x;                       // blk = b*N + i
  const float* row = sim + (size_t)blk * M;
  int t = threadIdx.x;
  float m = -3.402823466e38f;
  for (int j = t; j < M; j += 256) m = fmaxf(m, row[j]);
  red[t] = m; __syncthreads();
  for (int s = 128; s > 0; s >>= 1) { if (t < s) red[t] = fmaxf(red[t], red[t + s]); __syncthreads(); }
  float rm = red[0]; __syncthreads();
  float sum = 0.f;
  for (int j = t; j < M; j += 256) sum += __expf(row[j] - rm);
  red[t] = sum; __syncthreads();
  for (int s = 128; s > 0; s >>= 1) { if (t < s) red[t] += red[t + s]; __syncthreads(); }
  if (t == 0) { rmax[blk] = rm; rsum[blk] = red[0]; }
}

// ---------------- column softmax stats (32 cols/block, coalesced 8-row groups) ----------------
__global__ void k_colstats(const float* __restrict__ sim, float* __restrict__ cmax,
                           float* __restrict__ csum, int N, int M) {
  __shared__ float red[256];
  int b = blockIdx.y;
  int c = threadIdx.x & 31, rg = threadIdx.x >> 5;
  int col = blockIdx.x * 32 + c;
  const float* base = sim + (size_t)b * N * M;
  float m = -3.402823466e38f;
  for (int i = rg; i < N; i += 8) m = fmaxf(m, base[(size_t)i * M + col]);
  red[threadIdx.x] = m; __syncthreads();
  if (rg == 0) {
#pragma unroll
    for (int g = 1; g < 8; ++g) m = fmaxf(m, red[g * 32 + c]);
    red[c] = m;
  }
  __syncthreads();
  float cm = red[c]; __syncthreads();
  float sum = 0.f;
  for (int i = rg; i < N; i += 8) sum += __expf(base[(size_t)i * M + col] - cm);
  red[threadIdx.x] = sum; __syncthreads();
  if (rg == 0) {
#pragma unroll
    for (int g = 1; g < 8; ++g) sum += red[g * 32 + c];
    cmax[(size_t)b * M + col] = cm;
    csum[(size_t)b * M + col] = sum;
  }
}

// ---------------- per-row max of P ----------------
__global__ void k_rowpmax(const float* __restrict__ sim, const float* __restrict__ rmax,
                          const float* __restrict__ rsum, const float* __restrict__ cmax,
                          const float* __restrict__ csum, float* __restrict__ rpmax,
                          int N, int M) {
  __shared__ float red[256];
  int blk = blockIdx.x;
  int b = blk / N;
  const float* row = sim + (size_t)blk * M;
  float rm = rmax[blk], rs = rsum[blk];
  const float* cm = cmax + (size_t)b * M;
  const float* cs = csum + (size_t)b * M;
  int t = threadIdx.x;
  float pm = 0.f;
  for (int j = t; j < M; j += 256) pm = fmaxf(pm, dualP(row[j], rm, rs, cm[j], cs[j]));
  red[t] = pm; __syncthreads();
  for (int s = 128; s > 0; s >>= 1) { if (t < s) red[t] = fmaxf(red[t], red[t + s]); __syncthreads(); }
  if (t == 0) rpmax[blk] = red[0];
}

// ---------------- per-column max of P ----------------
__global__ void k_colpmax(const float* __restrict__ sim, const float* __restrict__ rmax,
                          const float* __restrict__ rsum, const float* __restrict__ cmax,
                          const float* __restrict__ csum, float* __restrict__ cpmax,
                          int N, int M) {
  __shared__ float red[256];
  int b = blockIdx.y;
  int c = threadIdx.x & 31, rg = threadIdx.x >> 5;
  int col = blockIdx.x * 32 + c;
  const float* base = sim + (size_t)b * N * M;
  float cm = cmax[(size_t)b * M + col], cs = csum[(size_t)b * M + col];
  float pm = 0.f;
  for (int i = rg; i < N; i += 8) {
    float s = base[(size_t)i * M + col];
    pm = fmaxf(pm, dualP(s, rmax[b * N + i], rsum[b * N + i], cm, cs));
  }
  red[threadIdx.x] = pm; __syncthreads();
  if (rg == 0) {
#pragma unroll
    for (int g = 1; g < 8; ++g) pm = fmaxf(pm, red[g * 32 + c]);
    cpmax[(size_t)b * M + col] = pm;
  }
}

// ---------------- mutual-NN count ----------------
__global__ void k_count(const float* __restrict__ sim, const float* __restrict__ rmax,
                        const float* __restrict__ rsum, const float* __restrict__ cmax,
                        const float* __restrict__ csum, const float* __restrict__ rpmax,
                        const float* __restrict__ cpmax, int* __restrict__ counts,
                        int rot, int N, int M) {
  __shared__ int red[256];
  int blk = blockIdx.x;
  int b = blk / N;
  const float* row = sim + (size_t)blk * M;
  float rm = rmax[blk], rs = rsum[blk], rp = rpmax[blk];
  const float* cm = cmax + (size_t)b * M;
  const float* cs = csum + (size_t)b * M;
  const float* cp = cpmax + (size_t)b * M;
  int t = threadIdx.x, cnt = 0;
  for (int j = t; j < M; j += 256) {
    float P = dualP(row[j], rm, rs, cm[j], cs[j]);
    if (P == rp && P == cp[j] && P > THRESHOLD) cnt++;
  }
  red[t] = cnt; __syncthreads();
  for (int s = 128; s > 0; s >>= 1) { if (t < s) red[t] += red[t + s]; __syncthreads(); }
  if (t == 0 && red[0] > 0) atomicAdd(&counts[rot], red[0]);
}

// ---------------- pick winning rotation (first strictly-greater wins, as in reference) ----------------
__global__ void k_select(const int* __restrict__ counts, int* __restrict__ bestRot,
                         float* __restrict__ outTail) {
  if (threadIdx.x == 0 && blockIdx.x == 0) {
    int best = counts[0], br = 0;
    for (int r = 1; r < 4; ++r)
      if (counts[r] > best) { best = counts[r]; br = r; }
    bestRot[0] = br;
    for (int r = 0; r < 4; ++r) outTail[r] = (float)counts[r];
    outTail[4] = (float)br;
  }
}

// ---------------- write masked best_P ----------------
__global__ void k_writeout(const float* __restrict__ sim, const float* __restrict__ rmaxA,
                           const float* __restrict__ rsumA, const float* __restrict__ rpmA,
                           const float* __restrict__ cmaxA, const float* __restrict__ csumA,
                           const float* __restrict__ cpmA, const int* __restrict__ bestRot,
                           float* __restrict__ out, int N, int M, int BN, int BM) {
  int br = bestRot[0];
  int blk = blockIdx.x;
  int b = blk / N;
  const float* row = sim + (size_t)blk * M;
  float rm = rmaxA[(size_t)br * BN + blk];
  float rs = rsumA[(size_t)br * BN + blk];
  float rp = rpmA [(size_t)br * BN + blk];
  const float* cm = cmaxA + (size_t)br * BM + (size_t)b * M;
  const float* cs = csumA + (size_t)br * BM + (size_t)b * M;
  const float* cp = cpmA  + (size_t)br * BM + (size_t)b * M;
  int t = threadIdx.x;
  for (int j = t; j < M; j += 256) {
    float P = dualP(row[j], rm, rs, cm[j], cs[j]);
    bool mut = (P == rp) && (P == cp[j]) && (P > THRESHOLD);
    out[(size_t)blk * M + j] = mut ? P : 0.f;
  }
}

__global__ void k_zero(int* __restrict__ counts) {
  if (threadIdx.x < 5) counts[threadIdx.x] = 0;
}

extern "C" void kernel_launch(void* const* d_in, const int* in_sizes, int n_in,
                              void* d_out, int out_size, void* d_ws, size_t ws_size,
                              hipStream_t stream) {
  const float* descA = (const float*)d_in[1];
  const float* descB = (const float*)d_in[3];
  const float* steer = (const float*)d_in[4];
  float* out = (float*)d_out;

  const int D  = D_DIM;
  const int BN = in_sizes[1] / D;   // 6144
  const int BM = in_sizes[3] / D;   // 6144
  const int Bb = 2;
  const int N  = BN / Bb;           // 3072
  const int M  = BM / Bb;           // 3072

  char* ws = (char*)d_ws;
  size_t off = 0;
  auto alloc = [&](size_t bytes) -> void* {
    void* p = ws + off;
    off += (bytes + 255) & ~(size_t)255;
    return p;
  };

  float*          sA       = (float*)alloc((size_t)BN * D * 4);
  float*          sB       = (float*)alloc((size_t)BN * D * 4);
  unsigned short* AnormAll = (unsigned short*)alloc((size_t)4 * BN * D * 2);
  unsigned short* Bnorm    = (unsigned short*)alloc((size_t)BM * D * 2);
  float*          sim      = (float*)alloc((size_t)BN * M * 4);   // 75.5 MB, L2-resident
  float*          rmaxA    = (float*)alloc((size_t)4 * BN * 4);
  float*          rsumA    = (float*)alloc((size_t)4 * BN * 4);
  float*          rpmA     = (float*)alloc((size_t)4 * BN * 4);
  float*          cmaxA    = (float*)alloc((size_t)4 * BM * 4);
  float*          csumA    = (float*)alloc((size_t)4 * BM * 4);
  float*          cpmA     = (float*)alloc((size_t)4 * BM * 4);
  int*            counts   = (int*)alloc(8 * sizeof(int));
  int*            bestRot  = counts + 4;

  dim3 blk(256);
  size_t slice = (size_t)BN * D;

  k_zero<<<dim3(1), blk, 0, stream>>>(counts);

  // Normalize B once; build all 4 steered+normalized A variants (fp32 steering chain).
  k_normalize<<<dim3(BM), blk, 0, stream>>>(descB, Bnorm);
  k_normalize<<<dim3(BN), blk, 0, stream>>>(descA, AnormAll + 0 * slice);
  k_steer    <<<dim3(BN), blk, 0, stream>>>(descA, steer, sA);
  k_normalize<<<dim3(BN), blk, 0, stream>>>(sA, AnormAll + 1 * slice);
  k_steer    <<<dim3(BN), blk, 0, stream>>>(sA, steer, sB);
  k_normalize<<<dim3(BN), blk, 0, stream>>>(sB, AnormAll + 2 * slice);
  k_steer    <<<dim3(BN), blk, 0, stream>>>(sB, steer, sA);
  k_normalize<<<dim3(BN), blk, 0, stream>>>(sA, AnormAll + 3 * slice);

  dim3 gGemm(M / 128, N / 128, Bb);
  dim3 gRow(BN);
  dim3 gCol(M / 32, Bb);

  for (int r = 0; r < 4; ++r) {
    k_simgemm <<<gGemm, blk, 0, stream>>>(AnormAll + r * slice, Bnorm, sim, N, M);
    k_rowstats<<<gRow,  blk, 0, stream>>>(sim, rmaxA + (size_t)r * BN, rsumA + (size_t)r * BN, N, M);
    k_colstats<<<gCol,  blk, 0, stream>>>(sim, cmaxA + (size_t)r * BM, csumA + (size_t)r * BM, N, M);
    k_rowpmax <<<gRow,  blk, 0, stream>>>(sim, rmaxA + (size_t)r * BN, rsumA + (size_t)r * BN,
                                          cmaxA + (size_t)r * BM, csumA + (size_t)r * BM,
                                          rpmA + (size_t)r * BN, N, M);
    k_colpmax <<<gCol,  blk, 0, stream>>>(sim, rmaxA + (size_t)r * BN, rsumA + (size_t)r * BN,
                                          cmaxA + (size_t)r * BM, csumA + (size_t)r * BM,
                                          cpmA + (size_t)r * BM, N, M);
    k_count   <<<gRow,  blk, 0, stream>>>(sim, rmaxA + (size_t)r * BN, rsumA + (size_t)r * BN,
                                          cmaxA + (size_t)r * BM, csumA + (size_t)r * BM,
                                          rpmA + (size_t)r * BN, cpmA + (size_t)r * BM,
                                          counts, r, N, M);
  }

  size_t BNM = (size_t)BN * M;
  k_select<<<dim3(1), dim3(32), 0, stream>>>(counts, bestRot, out + BNM);
  // Recompute sim only for the winning rotation, then write the masked output.
  k_simgemm_sel<<<gGemm, blk, 0, stream>>>(AnormAll, slice, bestRot, Bnorm, sim, N, M);
  k_writeout<<<gRow, blk, 0, stream>>>(sim, rmaxA, rsumA, rpmA, cmaxA, csumA, cpmA,
                                       bestRot, out, N, M, BN, BM);
}
